// SorscherRNN_28887950033605
// MI455X (gfx1250) — matrix operand order repaired and checked by reading, not compiled, not run on hardware
//
#include <hip/hip_runtime.h>

// Sorscher grid-cell RNN on MI455X (gfx1250), wave32 + v_wmma_f32_16x16x32_f16.
//
// Heavy math: 16x16x32 f16 WMMA, f32 accumulation.
// W_hh(f16)=32MB is L2-resident across all 64 recurrent steps (192MB L2).
// Each 512-thread block (16 waves) computes a 128x64 output tile; the 64
// B-columns are double-buffer staged through LDS so all 16 waves share them.
// Per stage: all 4 A fragments are fetched first (oldest loads -> partial
// loadcnt waits), then the next-stage B prefetch is issued (youngest load ->
// only waited at the ds_store point after compute).

typedef __attribute__((ext_vector_type(16))) _Float16 v16h;
typedef __attribute__((ext_vector_type(8)))  _Float16 v8h;
typedef __attribute__((ext_vector_type(4)))  _Float16 v4h;
typedef __attribute__((ext_vector_type(8)))  float    v8f;

#define B_  128
#define S_  64
#define NG_ 4096
#define NP_ 512

enum { EPI_H0 = 0, EPI_STEP = 1, EPI_OUT = 2 };

#define KC        128   // K-halves per LDS stage = 4 WMMA k-steps
#define BN        64    // block tile width  (output cols)
#define BM        128   // block tile height (output rows) = 8 M-tiles
#define NTHREADS  512   // 16 waves

// ---- A fragment: 16x32 f16 (M x K), contiguous k order, same as B frag ----
// lanes 0-15 : row M=lane,    VGPR0-3 = K 0..7,   VGPR4-7 = K 8..15
// lanes 16-31: row M=lane-16, VGPR0-3 = K 16..23, VGPR4-7 = K 24..31
__device__ __forceinline__ v16h load_a_frag(const _Float16* __restrict A, int lda,
                                            int m0, int k0, int lane) {
  const int hs = lane >> 4;
  const _Float16* p = A + (size_t)(m0 + (lane & 15)) * lda + k0 + hs * 16;
  const v8h lo = *(const v8h*)(p);        // global_load_b128
  const v8h hi = *(const v8h*)(p + 8);    // global_load_b128
  v16h r;
#pragma unroll
  for (int i = 0; i < 8; ++i) { r[i] = lo[i]; r[i + 8] = hi[i]; }
  return r;
}

// Block-tiled GEMM: C(BM x BN) = A(MxK,f16) * W(NxK,f16)^T, B staged in LDS.
// wave(warp) -> M-tile = warp>>1, 32-col half = warp&1, 2 accum tiles each.
// EPI_H0  : store f16 (encoder h0, no relu)
// EPI_STEP: h = relu(acc + v[b,t,0]*Wih[g,0] + v[b,t,1]*Wih[g,1]), store f16
// EPI_OUT : store f32 (decoder -> d_out)
template<int EPI>
__global__ void __launch_bounds__(NTHREADS)
gemm_lds(const _Float16* __restrict A, int lda,
         const _Float16* __restrict W, int ldw,
         int K,
         void* __restrict outp, int ldo,
         const float* __restrict v, const float* __restrict Wih, int t)
{
  __shared__ alignas(256) _Float16 sB[2][BN][KC];   // 32 KB, double-buffered

  const int tid  = threadIdx.x;
  const int lane = tid & 31;
  const int warp = tid >> 5;
  const int mt   = warp >> 1;                  // 0..7
  const int nh   = warp & 1;                   // 0..1
  const int n0   = blockIdx.x * BN;
  const int m0   = blockIdx.y * BM + mt * 16;

  // cooperative B staging: each thread owns one 32B chunk per stage
  // (64 rows x 256B per stage = 512 x 32B)
  const int crow  = (tid * 2) >> 4;            // 0..63 : B row (= output col)
  const int choff = ((tid * 2) & 15) * 8;      // half offset 0..112 (32B aligned)
  const _Float16* gB = W + (size_t)(n0 + crow) * ldw + choff;

  const int nStages = K / KC;

  v8f acc[2] = {};

  // prologue: fetch + commit stage 0
  v16h pre = *(const v16h*)(gB);
  *(v16h*)&sB[0][crow][choff] = pre;
  __syncthreads();

  for (int s = 0; s < nStages; ++s) {
    const int ks  = s * KC;
    const int buf = s & 1;

    // 1) all A fragments of this stage first -> oldest outstanding loads,
    //    enables partial s_wait_loadcnt per k-step instead of full drains
    v16h af[KC / 32];
#pragma unroll
    for (int kk = 0; kk < KC / 32; ++kk)
      af[kk] = load_a_frag(A, lda, m0, ks + kk * 32, lane);

    // 2) next stage's B prefetch AFTER the A loads -> youngest load, only
    //    waited at the ds_store after the compute phase
    if (s + 1 < nStages)
      pre = *(const v16h*)(gB + (size_t)(s + 1) * KC);

    // 3) compute: 4 k-steps x 2 tiles, B fragments from LDS
#pragma unroll
    for (int kk = 0; kk < KC / 32; ++kk) {
#pragma unroll
      for (int j = 0; j < 2; ++j) {
        // B 32x16 frag: lanes 0-15 K 0..15, lanes 16-31 K 16..31
        const int cb = nh * 32 + j * 16 + (lane & 15);
        const v16h b = *(const v16h*)&sB[buf][cb][kk * 32 + (lane >> 4) * 16];
        acc[j] = __builtin_amdgcn_wmma_f32_16x16x32_f16(
            false, af[kk], false, b, (short)0, acc[j], false, false);
      }
    }

    __syncthreads();                           // everyone done reading buf^1
    if (s + 1 < nStages)
      *(v16h*)&sB[(s + 1) & 1][crow][choff] = pre;
    __syncthreads();                           // next stage visible
  }

  // C/D layout: VGPR r -> lanes 0-15: (M=r, N=lane); lanes 16-31: (M=8+r, N=lane-16)
  const int col   = lane & 15;
  const int rbase = (lane >> 4) * 8;

  if constexpr (EPI == EPI_STEP) {
    _Float16* __restrict out = (_Float16*)outp;
    float v0[8], v1[8];
#pragma unroll
    for (int r = 0; r < 8; ++r) {
      const int b = m0 + rbase + r;            // M dimension == batch
      v0[r] = v[(size_t)(b * S_ + t) * 2 + 0];
      v1[r] = v[(size_t)(b * S_ + t) * 2 + 1];
    }
#pragma unroll
    for (int j = 0; j < 2; ++j) {
      const int g   = n0 + nh * 32 + j * 16 + col;
      const float wi0 = Wih[g * 2 + 0];
      const float wi1 = Wih[g * 2 + 1];
#pragma unroll
      for (int r = 0; r < 8; ++r) {
        float h = fmaxf(acc[j][r] + v0[r] * wi0 + v1[r] * wi1, 0.0f);
        out[(size_t)(m0 + rbase + r) * ldo + g] = (_Float16)h;
      }
    }
  } else if constexpr (EPI == EPI_H0) {
    _Float16* __restrict out = (_Float16*)outp;
#pragma unroll
    for (int j = 0; j < 2; ++j) {
      const int n = n0 + nh * 32 + j * 16 + col;
#pragma unroll
      for (int r = 0; r < 8; ++r)
        out[(size_t)(m0 + rbase + r) * ldo + n] = (_Float16)acc[j][r];
    }
  } else {
    float* __restrict out = (float*)outp;
#pragma unroll
    for (int j = 0; j < 2; ++j) {
      const int n = n0 + nh * 32 + j * 16 + col;
#pragma unroll
      for (int r = 0; r < 8; ++r)
        out[(size_t)(m0 + rbase + r) * ldo + n] = acc[j][r];
    }
  }
}

// one-shot f32 -> f16 conversion, 4 elements per thread
__global__ void cvt_f32_to_f16(const float* __restrict s, _Float16* __restrict d, int n4) {
  const int i = blockIdx.x * blockDim.x + threadIdx.x;
  if (i < n4) {
    const float4 f = ((const float4*)s)[i];
    v4h h;
    h[0] = (_Float16)f.x; h[1] = (_Float16)f.y;
    h[2] = (_Float16)f.z; h[3] = (_Float16)f.w;
    ((v4h*)d)[i] = h;
  }
}

extern "C" void kernel_launch(void* const* d_in, const int* in_sizes, int n_in,
                              void* d_out, int out_size, void* d_ws, size_t ws_size,
                              hipStream_t stream) {
  (void)in_sizes; (void)n_in; (void)out_size; (void)ws_size;
  const float* v    = (const float*)d_in[0];   // (B,S,2)
  const float* p0   = (const float*)d_in[1];   // (B,NP)
  const float* Wenc = (const float*)d_in[2];   // (NG,NP)
  const float* Wih  = (const float*)d_in[3];   // (NG,2)  kept f32 (epilogue)
  const float* Whh  = (const float*)d_in[4];   // (NG,NG)
  const float* Wdec = (const float*)d_in[5];   // (NP,NG)
  float* out = (float*)d_out;                  // (B,S,NP) f32

  // workspace layout (all regions fully overwritten before use)
  char* ws = (char*)d_ws;
  _Float16* Whh_h  = (_Float16*)(ws);                                     // 32 MB
  _Float16* Wenc_h = (_Float16*)(ws + (size_t)32 * 1024 * 1024);          //  4 MB
  _Float16* Wdec_h = (_Float16*)(ws + (size_t)36 * 1024 * 1024);          //  4 MB
  _Float16* p0_h   = (_Float16*)(ws + (size_t)40 * 1024 * 1024);          // 128 KB
  _Float16* H0     = (_Float16*)(ws + (size_t)40 * 1024 * 1024 + 131072); //  1 MB
  _Float16* HS     = (_Float16*)(ws + (size_t)42 * 1024 * 1024);          // 64 MB [b][t][g]

  auto cvt = [&](const float* s, _Float16* d, int n) {
    const int n4 = n >> 2;
    cvt_f32_to_f16<<<(n4 + 255) / 256, 256, 0, stream>>>(s, d, n4);
  };
  cvt(Whh,  Whh_h,  NG_ * NG_);
  cvt(Wenc, Wenc_h, NG_ * NP_);
  cvt(Wdec, Wdec_h, NP_ * NG_);
  cvt(p0,   p0_h,   B_  * NP_);

  // h0 = p0 @ Wenc^T : 128x4096, K=512 -> 64 blocks of 128x64
  gemm_lds<EPI_H0><<<dim3(64, 1), NTHREADS, 0, stream>>>(
      p0_h, NP_, Wenc_h, NP_, NP_, H0, NG_, nullptr, nullptr, 0);

  // 64 recurrent steps: h_t = relu(vi_t + h_{t-1} @ Whh^T) -> HS[b][t][g]
  for (int t = 0; t < S_; ++t) {
    const _Float16* Aprev = (t == 0) ? H0 : (HS + (size_t)(t - 1) * NG_);
    const int lda         = (t == 0) ? NG_ : (S_ * NG_);
    _Float16* Hout = HS + (size_t)t * NG_;     // row b lands at (b*S + t)*NG
    gemm_lds<EPI_STEP><<<dim3(64, 1), NTHREADS, 0, stream>>>(
        Aprev, lda, Whh_h, NG_, NG_, Hout, S_ * NG_, v, Wih, t);
  }

  // out = HS @ Wdec^T : M=8192, N=512, K=4096 -> 8 x 64 blocks of 128x64
  gemm_lds<EPI_OUT><<<dim3(8, 64), NTHREADS, 0, stream>>>(
      HS, NG_, Wdec_h, NG_, NG_, out, NP_, nullptr, nullptr, 0);
}
